// Net_87445534146910
// MI455X (gfx1250) — compile-verified, matrix operand order checked by reading, not run on hardware
//
#include <hip/hip_runtime.h>
#include <hip/hip_bf16.h>

// ---------------------------------------------------------------------------
// Net dims (fixed by the reference)
#define Bn  64
#define Nv  512
#define Fn  44
#define Dn  512
#define Hn  2048
#define Lnum 2
#define BNn (Bn * Nv)

typedef unsigned short u16;
typedef long long ll;
typedef __attribute__((ext_vector_type(16))) __bf16 v16bf;
typedef __attribute__((ext_vector_type(8)))  float  v8f;

__device__ __forceinline__ float bf2f(u16 u) {
    union { unsigned int i; float f; } c; c.i = ((unsigned int)u) << 16; return c.f;
}
__device__ __forceinline__ u16 f2bf(float f) {
    union { float f; unsigned int i; } c; c.f = f;
    unsigned int x = c.i;
    return (u16)((x + 0x7FFFu + ((x >> 16) & 1u)) >> 16);   // round-to-nearest-even
}

// A 16x32 bf16 fragment (ISA 7.12.2): per lane, elements 0..7 = K[kk+8*lh .. +7],
// elements 8..15 = K[kk+16+8*lh .. +7]  ->  two contiguous 16B chunks.
union FragU {
    uint4 q[2];
    v16bf v;
};

// ---------------------------------------------------------------------------
// Generic batched GEMM, ALL operands K-contiguous:
//   A[b]: M rows, row stride a_rs (elements), K contiguous
//   B[b]: N cols, col stride b_ns (elements), K contiguous ("[N,K]" layout)
// C[b] = act( (A@B + bias) * smul * emul * colscale + resid + (accum?Cb_old:0) )
// fp32 accumulate via v_wmma_f32_16x16x32_bf16. Block = 256 thr = 8 wave32;
// block tile 256(M) x 64(N); each wave owns a 32x64 strip:
//   2 A fragments x 4 B fragments -> 8 WMMAs per K-step, 6 fragment loads.
// All dims are multiples of 256/64/32 -> no edge handling.
__global__ __launch_bounds__(256)
void gemm_bf16_wmma(const u16* __restrict__ A,  ll a_rs, ll a_bs,
                    const u16* __restrict__ Bm, ll b_ns, ll b_bs,
                    const float* __restrict__ bias, const float* __restrict__ colscale,
                    const float* __restrict__ resid, ll r_rs, ll r_bs,
                    const float* __restrict__ emul,  ll e_rs, ll e_bs,
                    float smul, int act,
                    float* __restrict__ Cf, ll c_rs, ll c_ns, ll c_bs,
                    u16*   __restrict__ Cb, ll cb_rs, ll cb_ns, ll cb_bs, int accum,
                    int K)
{
    const int lane = threadIdx.x & 31;
    const int wave = threadIdx.x >> 5;
    const int lh   = lane >> 4;     // lane half (fragment layout)
    const int lm   = lane & 15;
    const ll  bz   = blockIdx.z;
    const ll  m0   = (ll)blockIdx.x * 256 + (ll)wave * 32;
    const ll  n0   = (ll)blockIdx.y * 64;

    // 16B-aligned chunk pointers (all strides are multiples of 8 elements)
    const uint4* A4a  = (const uint4*)(A + bz * a_bs + (m0 +  0 + lm) * a_rs);
    const uint4* A4b  = (const uint4*)(A + bz * a_bs + (m0 + 16 + lm) * a_rs);
    const uint4* B4_0 = (const uint4*)(Bm + bz * b_bs + (n0 +  0 + lm) * b_ns);
    const uint4* B4_1 = (const uint4*)(Bm + bz * b_bs + (n0 + 16 + lm) * b_ns);
    const uint4* B4_2 = (const uint4*)(Bm + bz * b_bs + (n0 + 32 + lm) * b_ns);
    const uint4* B4_3 = (const uint4*)(Bm + bz * b_bs + (n0 + 48 + lm) * b_ns);

    v8f zero = {};
    v8f a00 = zero, a01 = zero, a02 = zero, a03 = zero;   // rows m0..m0+15
    v8f a10 = zero, a11 = zero, a12 = zero, a13 = zero;   // rows m0+16..m0+31

#pragma unroll 2
    for (int kk = 0; kk < K; kk += 32) {
        const int idx = (kk >> 3) + lh;          // uint4 units (8 bf16 each)
        FragU fa0, fa1, f0, f1, f2, f3;
        fa0.q[0] = A4a[idx];  fa0.q[1] = A4a[idx + 2];
        fa1.q[0] = A4b[idx];  fa1.q[1] = A4b[idx + 2];
        f0.q[0] = B4_0[idx];  f0.q[1] = B4_0[idx + 2];
        f1.q[0] = B4_1[idx];  f1.q[1] = B4_1[idx + 2];
        f2.q[0] = B4_2[idx];  f2.q[1] = B4_2[idx + 2];
        f3.q[0] = B4_3[idx];  f3.q[1] = B4_3[idx + 2];
        a00 = __builtin_amdgcn_wmma_f32_16x16x32_bf16(false, fa0.v, false, f0.v, (short)0, a00, false, false);
        a10 = __builtin_amdgcn_wmma_f32_16x16x32_bf16(false, fa1.v, false, f0.v, (short)0, a10, false, false);
        a01 = __builtin_amdgcn_wmma_f32_16x16x32_bf16(false, fa0.v, false, f1.v, (short)0, a01, false, false);
        a11 = __builtin_amdgcn_wmma_f32_16x16x32_bf16(false, fa1.v, false, f1.v, (short)0, a11, false, false);
        a02 = __builtin_amdgcn_wmma_f32_16x16x32_bf16(false, fa0.v, false, f2.v, (short)0, a02, false, false);
        a12 = __builtin_amdgcn_wmma_f32_16x16x32_bf16(false, fa1.v, false, f2.v, (short)0, a12, false, false);
        a03 = __builtin_amdgcn_wmma_f32_16x16x32_bf16(false, fa0.v, false, f3.v, (short)0, a03, false, false);
        a13 = __builtin_amdgcn_wmma_f32_16x16x32_bf16(false, fa1.v, false, f3.v, (short)0, a13, false, false);
    }

    v8f accs[2][4] = {{a00, a01, a02, a03}, {a10, a11, a12, a13}};
    // Epilogue. C/D layout: VGPR r -> row m0 + a*16 + r + 8*lh, col n0 + jf*16 + lm.
#pragma unroll
    for (int jf = 0; jf < 4; ++jf) {
        const ll    n  = n0 + jf * 16 + lm;
        const float bi = bias     ? bias[n]     : 0.f;
        const float cs = colscale ? colscale[n] : 1.f;
#pragma unroll
        for (int a = 0; a < 2; ++a) {
#pragma unroll
            for (int r = 0; r < 8; ++r) {
                const ll m = m0 + a * 16 + r + 8 * lh;
                float v = accs[a][jf][r] + bi;
                v *= smul;
                if (emul)          v *= emul[bz * e_bs + m * e_rs + n];
                v *= cs;
                if (resid)         v += resid[bz * r_bs + m * r_rs + n];
                if (Cb && accum)   v += bf2f(Cb[bz * cb_bs + m * cb_rs + n * cb_ns]);
                if (act == 1)      v = fmaxf(v, 0.f);
                else if (act == 2) v = 1.f / (1.f + expf(-v));
                if (Cf) Cf[bz * c_bs + m * c_rs + n * c_ns]     = v;
                if (Cb) Cb[bz * cb_bs + m * cb_rs + n * cb_ns]  = f2bf(v);
            }
        }
    }
}

// ---------------------------------------------------------------------------
__global__ void cast_f2bf_kernel(const float* __restrict__ src, u16* __restrict__ dst, ll n) {
    ll i = (ll)blockIdx.x * blockDim.x + threadIdx.x;
    if (i < n) dst[i] = f2bf(src[i]);
}

// fp32 [K,N] row-major  ->  bf16 [N,K] row-major (transpose-cast)
__global__ void transcast_kernel(const float* __restrict__ src, u16* __restrict__ dst,
                                 int K, int N) {
    ll i = (ll)blockIdx.x * blockDim.x + threadIdx.x;
    if (i >= (ll)K * N) return;
    const ll k = i / N, n = i % N;
    dst[n * (ll)K + k] = f2bf(src[i]);
}

// bf16 [B,N,D] -> bf16 [B,D,N]
__global__ void transbf_kernel(const u16* __restrict__ src, u16* __restrict__ dst) {
    ll i = (ll)blockIdx.x * blockDim.x + threadIdx.x;
    if (i >= (ll)Bn * Nv * Dn) return;
    const ll b = i / ((ll)Nv * Dn);
    const ll rem = i % ((ll)Nv * Dn);
    const ll n = rem / Dn, d = rem % Dn;
    dst[b * (ll)Dn * Nv + d * Nv + n] = src[i];
}

// adj fp32 [B,N,N] -> bf16 adj and bf16 adj^T
__global__ void adjcast_kernel(const float* __restrict__ adj,
                               u16* __restrict__ a, u16* __restrict__ at) {
    ll i = (ll)blockIdx.x * blockDim.x + threadIdx.x;
    if (i >= (ll)Bn * Nv * Nv) return;
    const ll b = i / ((ll)Nv * Nv);
    const ll rem = i % ((ll)Nv * Nv);
    const ll m = rem / Nv, n = rem % Nv;
    const u16 v = f2bf(adj[i]);
    a[i] = v;
    at[b * (ll)Nv * Nv + n * Nv + m] = v;
}

__global__ void relu_bf_kernel(u16* __restrict__ p, ll n) {
    ll i = (ll)blockIdx.x * blockDim.x + threadIdx.x;
    if (i < n) p[i] = f2bf(fmaxf(bf2f(p[i]), 0.f));
}

// h = x @ emb_w + emb_b  (K=44, scalar fp32; also writes bf16 copy)
__global__ void embed_kernel(const float* __restrict__ x, const float* __restrict__ W,
                             const float* __restrict__ b,
                             float* __restrict__ h, u16* __restrict__ hbf) {
    ll i = (ll)blockIdx.x * blockDim.x + threadIdx.x;
    if (i >= (ll)BNn * Dn) return;
    const ll  row = i / Dn;
    const int d   = (int)(i % Dn);
    float acc = b[d];
#pragma unroll 4
    for (int f = 0; f < Fn; ++f) acc += x[row * Fn + f] * W[(ll)f * Dn + d];
    h[i]   = acc;
    hbf[i] = f2bf(acc);
}

// LayerNorm over last dim (D=512): one block (256 thr) per row.
__global__ __launch_bounds__(256)
void ln_kernel(const float* __restrict__ h, const float* __restrict__ g,
               const float* __restrict__ bta, float scale,
               u16* __restrict__ obf, float* __restrict__ of32) {
    const int t   = threadIdx.x;
    const ll  row = blockIdx.x;
    const float* hr = h + row * (ll)Dn;
    const float x0 = hr[t], x1 = hr[t + 256];
    __shared__ float s[256];
    s[t] = x0 + x1; __syncthreads();
    for (int o = 128; o > 0; o >>= 1) { if (t < o) s[t] += s[t + o]; __syncthreads(); }
    const float mean = s[0] * (1.f / Dn);
    __syncthreads();
    const float d0 = x0 - mean, d1 = x1 - mean;
    s[t] = d0 * d0 + d1 * d1; __syncthreads();
    for (int o = 128; o > 0; o >>= 1) { if (t < o) s[t] += s[t + o]; __syncthreads(); }
    const float rstd = rsqrtf(s[0] * (1.f / Dn) + 1e-5f);
    const float y0 = (d0 * rstd * g[t]       + bta[t])       * scale;
    const float y1 = (d1 * rstd * g[t + 256] + bta[t + 256]) * scale;
    obf[row * Dn + t]       = f2bf(y0);
    obf[row * Dn + t + 256] = f2bf(y1);
    if (of32) { of32[row * Dn + t] = y0; of32[row * Dn + t + 256] = y1; }
}

// attn = scores / (rowsum + 1e-6), stored bf16. One block per (b,n) row.
__global__ __launch_bounds__(256)
void attnnorm_kernel(const float* __restrict__ scores, u16* __restrict__ attnbf) {
    const int t   = threadIdx.x;
    const ll  row = blockIdx.x;
    const float* sr = scores + row * (ll)Nv;
    const float x0 = sr[t], x1 = sr[t + 256];
    __shared__ float s[256];
    s[t] = x0 + x1; __syncthreads();
    for (int o = 128; o > 0; o >>= 1) { if (t < o) s[t] += s[t + o]; __syncthreads(); }
    const float inv = 1.f / (s[0] + 1e-6f);
    attnbf[row * Nv + t]       = f2bf(x0 * inv);
    attnbf[row * Nv + t + 256] = f2bf(x1 * inv);
}

// hsum[b,d] = sum_n h[b,n,d]
__global__ void nodesum_kernel(const float* __restrict__ h, float* __restrict__ hsum) {
    int i = blockIdx.x * blockDim.x + threadIdx.x;
    if (i >= Bn * Dn) return;
    const int b = i / Dn, d = i % Dn;
    float s = 0.f;
    for (int n = 0; n < Nv; ++n) s += h[((ll)b * Nv + n) * Dn + d];
    hsum[i] = s;
}

// Predictor head: z=[hsum,sf] -> relu(z@w1+b1) @ w2 + b2 -> softplus. Block per b.
__global__ __launch_bounds__(256)
void head_kernel(const float* __restrict__ hsum, const float* __restrict__ sf,
                 const float* __restrict__ w1, const float* __restrict__ b1,
                 const float* __restrict__ w2, const float* __restrict__ b2,
                 float* __restrict__ out) {
    const int t = threadIdx.x, b = blockIdx.x;
    float partial = 0.f;
    for (int d = t; d < Dn; d += 256) {
        float acc = b1[d];
        for (int k = 0; k < Dn; ++k) acc += hsum[b * Dn + k] * w1[(ll)k * Dn + d];
#pragma unroll
        for (int k = 0; k < 4; ++k)  acc += sf[b * 4 + k] * w1[(ll)(Dn + k) * Dn + d];
        partial += fmaxf(acc, 0.f) * w2[d];
    }
    __shared__ float s[256];
    s[t] = partial; __syncthreads();
    for (int o = 128; o > 0; o >>= 1) { if (t < o) s[t] += s[t + o]; __syncthreads(); }
    if (t == 0) {
        const float v = s[0] + b2[0];
        out[b] = fmaxf(v, 0.f) + log1pf(expf(-fabsf(v)));   // stable softplus
    }
}

// ---------------------------------------------------------------------------
extern "C" void kernel_launch(void* const* d_in, const int* in_sizes, int n_in,
                              void* d_out, int out_size, void* d_ws, size_t ws_size,
                              hipStream_t stream) {
    const float* x      = (const float*)d_in[0];
    const float* adj    = (const float*)d_in[1];
    const float* sf     = (const float*)d_in[2];
    const float* emb_w  = (const float*)d_in[3];
    const float* emb_b  = (const float*)d_in[4];
    const float* qk_w   = (const float*)d_in[5];
    const float* qk_b   = (const float*)d_in[6];
    const float* lnl_w  = (const float*)d_in[7];
    const float* lnl_b  = (const float*)d_in[8];
    const float* lnr_w  = (const float*)d_in[9];
    const float* ffn_g  = (const float*)d_in[10];
    const float* ffn_b  = (const float*)d_in[11];
    const float* fc1_w  = (const float*)d_in[12];
    const float* fc1_b  = (const float*)d_in[13];
    const float* gcn_w  = (const float*)d_in[14];
    const float* gcn_b  = (const float*)d_in[15];
    const float* fc2_w  = (const float*)d_in[16];
    const float* fc2_b  = (const float*)d_in[17];
    const float* ls     = (const float*)d_in[18];
    const float* pr_g   = (const float*)d_in[19];
    const float* pr_b   = (const float*)d_in[20];
    const float* pr_w1  = (const float*)d_in[21];
    const float* pr_b1  = (const float*)d_in[22];
    const float* pr_w2  = (const float*)d_in[23];
    const float* pr_b2  = (const float*)d_in[24];
    const float* pd_w1  = (const float*)d_in[25];
    const float* pd_b1  = (const float*)d_in[26];
    const float* pd_w2  = (const float*)d_in[27];
    const float* pd_b2  = (const float*)d_in[28];
    float* outp = (float*)d_out;

    // ---- workspace carve-out ----
    char* ws = (char*)d_ws; size_t off = 0;
    auto alloc = [&](size_t bytes) -> void* {
        void* p = ws + off; off += (bytes + 255) & ~(size_t)255; return p;
    };
    float* h       = (float*)alloc((size_t)BNn * Dn * 4);
    u16*   hbf     = (u16*)  alloc((size_t)BNn * Dn * 2);
    u16*   hTbf    = (u16*)  alloc((size_t)BNn * Dn * 2);     // [B, D, N]
    u16*   hnbf    = (u16*)  alloc((size_t)BNn * Dn * 2);
    u16*   qkbf    = (u16*)  alloc((size_t)BNn * Dn * 2);     // also reused as out_bf
    u16*   attnbf  = (u16*)  alloc((size_t)Bn * Nv * Nv * 2);
    u16*   adjbf   = (u16*)  alloc((size_t)Bn * Nv * Nv * 2);
    u16*   adjTbf  = (u16*)  alloc((size_t)Bn * Nv * Nv * 2);
    float* scoresF = (float*)alloc((size_t)Bn * Nv * Nv * 4); // reused as fp32 [BN,D] scratch
    u16*   gT      = (u16*)  alloc((size_t)BNn * Hn * 2);     // [B, H, N]
    u16*   vbf     = (u16*)  alloc((size_t)BNn * Hn * 2);     // [BN, H]
    u16*   qk_wT   = (u16*)  alloc((size_t)Lnum * Dn * Dn * 2);
    u16*   lnl_wT  = (u16*)  alloc((size_t)Lnum * Dn * Dn * 2);
    u16*   lnr_wT  = (u16*)  alloc((size_t)Lnum * Dn * Dn * 2);
    u16*   gcn_wT  = (u16*)  alloc((size_t)Lnum * Hn * Dn * 2);
    u16*   fc1_wT  = (u16*)  alloc((size_t)Lnum * Hn * Dn * 2);
    u16*   fc2_wT  = (u16*)  alloc((size_t)Lnum * Dn * Hn * 2);
    u16*   pr_w1T  = (u16*)  alloc((size_t)2 * Dn * Dn * 2);
    u16*   pr_w2T  = (u16*)  alloc((size_t)Dn * 2 * Dn * 2);
    float* hsum    = (float*)alloc((size_t)Bn * Dn * 4);
    (void)ws_size; (void)in_sizes; (void)n_in; (void)out_size;

    auto transcast = [&](const float* s, u16* d, int K, int N) {
        transcast_kernel<<<(unsigned)(((ll)K * N + 255) / 256), 256, 0, stream>>>(s, d, K, N);
    };
    auto gemm = [&](const u16* A, ll a_rs, ll a_bs,
                    const u16* Bm, ll b_ns, ll b_bs,
                    const float* bias, const float* colscale,
                    const float* resid, ll r_rs, ll r_bs,
                    const float* emul, ll e_rs, ll e_bs,
                    float smul, int act,
                    float* Cf, ll c_rs, ll c_ns, ll c_bs,
                    u16* Cb, ll cb_rs, ll cb_ns, ll cb_bs, int accum,
                    int M, int Ncols, int K, int batch) {
        dim3 grid(M / 256, Ncols / 64, batch);
        gemm_bf16_wmma<<<grid, 256, 0, stream>>>(A, a_rs, a_bs, Bm, b_ns, b_bs,
                                                 bias, colscale, resid, r_rs, r_bs,
                                                 emul, e_rs, e_bs, smul, act,
                                                 Cf, c_rs, c_ns, c_bs,
                                                 Cb, cb_rs, cb_ns, cb_bs, accum, K);
    };

    // ---- prep: transposed bf16 weights, adjacency, embedding ----
    for (int i = 0; i < Lnum; ++i) {
        transcast(qk_w  + (size_t)i * Dn * Dn, qk_wT  + (size_t)i * Dn * Dn, Dn, Dn);
        transcast(lnl_w + (size_t)i * Dn * Dn, lnl_wT + (size_t)i * Dn * Dn, Dn, Dn);
        transcast(lnr_w + (size_t)i * Dn * Dn, lnr_wT + (size_t)i * Dn * Dn, Dn, Dn);
        transcast(gcn_w + (size_t)i * Dn * Hn, gcn_wT + (size_t)i * Hn * Dn, Dn, Hn);
        transcast(fc1_w + (size_t)i * Dn * Hn, fc1_wT + (size_t)i * Hn * Dn, Dn, Hn);
        transcast(fc2_w + (size_t)i * Hn * Dn, fc2_wT + (size_t)i * Dn * Hn, Hn, Dn);
    }
    transcast(pr_w1, pr_w1T, Dn, 2 * Dn);
    transcast(pr_w2, pr_w2T, 2 * Dn, Dn);
    adjcast_kernel<<<(unsigned)(((ll)Bn * Nv * Nv + 255) / 256), 256, 0, stream>>>(adj, adjbf, adjTbf);
    embed_kernel<<<(unsigned)(((ll)BNn * Dn + 255) / 256), 256, 0, stream>>>(x, emb_w, emb_b, h, hbf);
    transbf_kernel<<<(unsigned)(((ll)BNn * Dn + 255) / 256), 256, 0, stream>>>(hbf, hTbf);

    const float inv_sqrt_d = 0.044194173824159216f;  // 1/sqrt(512)

    for (int i = 0; i < Lnum; ++i) {
        const u16* qkW  = qk_wT  + (size_t)i * Dn * Dn;
        const u16* lnlW = lnl_wT + (size_t)i * Dn * Dn;
        const u16* lnrW = lnr_wT + (size_t)i * Dn * Dn;
        const u16* gcnW = gcn_wT + (size_t)i * Hn * Dn;
        const u16* fc1W = fc1_wT + (size_t)i * Hn * Dn;
        const u16* fc2W = fc2_wT + (size_t)i * Dn * Hn;

        // QK = sigmoid(h @ qk_w + qk_b)  -> bf16
        gemm(hbf, Dn, 0, qkW, Dn, 0, qk_b + (size_t)i * Dn, nullptr,
             nullptr, 0, 0, nullptr, 0, 0, 1.f, 2,
             nullptr, 0, 0, 0, qkbf, Dn, 1, 0, 0, BNn, Dn, Dn, 1);
        // scores = (QK @ QK^T) * inv * adj   (B = QK row-major: K contiguous)
        gemm(qkbf, Dn, (ll)Nv * Dn, qkbf, Dn, (ll)Nv * Dn, nullptr, nullptr,
             nullptr, 0, 0, adj, Nv, (ll)Nv * Nv, inv_sqrt_d, 0,
             scoresF, Nv, 1, (ll)Nv * Nv, nullptr, 0, 0, 0, 0, Nv, Nv, Dn, Bn);
        // row-normalize -> bf16 attn
        attnnorm_kernel<<<Bn * Nv, 256, 0, stream>>>(scoresF, attnbf);
        // out = attn @ h  (B = hT, K contiguous)  -> bf16 (reuse qkbf)
        gemm(attnbf, Nv, (ll)Nv * Nv, hTbf, Nv, (ll)Dn * Nv, nullptr, nullptr,
             nullptr, 0, 0, nullptr, 0, 0, 1.f, 0,
             nullptr, 0, 0, 0, qkbf, Dn, 1, (ll)Nv * Dn, 0, Nv, Dn, Dn, Bn);
        // C1 = out @ lnl_w + lnl_b  (fp32, into scoresF viewed as [BN,D])
        gemm(qkbf, Dn, 0, lnlW, Dn, 0, lnl_b + (size_t)i * Dn, nullptr,
             nullptr, 0, 0, nullptr, 0, 0, 1.f, 0,
             scoresF, Dn, 1, 0, nullptr, 0, 0, 0, 0, BNn, Dn, Dn, 1);
        // h = relu(h @ lnr_w + C1)
        gemm(hbf, Dn, 0, lnrW, Dn, 0, nullptr, nullptr,
             scoresF, Dn, 0, nullptr, 0, 0, 1.f, 1,
             h, Dn, 1, 0, nullptr, 0, 0, 0, 0, BNn, Dn, Dn, 1);
        // hn = LN(h)  -> bf16
        ln_kernel<<<BNn, 256, 0, stream>>>(h, ffn_g + (size_t)i * Dn, ffn_b + (size_t)i * Dn,
                                           1.f, hnbf, nullptr);
        // gT = (hn @ gcn_w + gcn_b)^T  per batch -> [B, H, N] (transposed store)
        gemm(hnbf, Dn, (ll)Nv * Dn, gcnW, Dn, 0, gcn_b + (size_t)i * Hn, nullptr,
             nullptr, 0, 0, nullptr, 0, 0, 1.f, 0,
             nullptr, 0, 0, 0, gT, 1, Nv, (ll)Hn * Nv, 0, Nv, Hn, Dn, Bn);
        // v = hn @ fc1_w + fc1_b  -> bf16 [BN, H]
        gemm(hnbf, Dn, 0, fc1W, Dn, 0, fc1_b + (size_t)i * Hn, nullptr,
             nullptr, 0, 0, nullptr, 0, 0, 1.f, 0,
             nullptr, 0, 0, 0, vbf, Hn, 1, 0, 0, BNn, Hn, Dn, 1);
        // v[:, :H/2] += adj @ g1     (B = gT rows 0..H/2, K contiguous)
        gemm(adjbf, Nv, (ll)Nv * Nv, gT, Nv, (ll)Hn * Nv, nullptr, nullptr,
             nullptr, 0, 0, nullptr, 0, 0, 1.f, 0,
             nullptr, 0, 0, 0, vbf, Hn, 1, (ll)Nv * Hn, 1, Nv, Hn / 2, Nv, Bn);
        // v[:, H/2:] += adj^T @ g2   (A = adjT, K contiguous)
        gemm(adjTbf, Nv, (ll)Nv * Nv, gT + (ll)(Hn / 2) * Nv, Nv, (ll)Hn * Nv, nullptr, nullptr,
             nullptr, 0, 0, nullptr, 0, 0, 1.f, 0,
             nullptr, 0, 0, 0, vbf + Hn / 2, Hn, 1, (ll)Nv * Hn, 1, Nv, Hn / 2, Nv, Bn);
        // v = relu(v)
        relu_bf_kernel<<<(unsigned)(((ll)BNn * Hn + 255) / 256), 256, 0, stream>>>(vbf, (ll)BNn * Hn);
        // h = h + (v @ fc2_w + fc2_b) * ls   (refresh hbf, then hT)
        gemm(vbf, Hn, 0, fc2W, Hn, 0, fc2_b + (size_t)i * Dn, ls + (size_t)i * Dn,
             h, Dn, 0, nullptr, 0, 0, 1.f, 0,
             h, Dn, 1, 0, hbf, Dn, 1, 0, 0, BNn, Dn, Hn, 1);
        transbf_kernel<<<(unsigned)(((ll)BNn * Dn + 255) / 256), 256, 0, stream>>>(hbf, hTbf);
    }

    // ---- PreReduction: hn = LN(h)*0.01 (bf16 + fp32 copy in scoresF) ----
    ln_kernel<<<BNn, 256, 0, stream>>>(h, pr_g, pr_b, 0.01f, hnbf, scoresF);
    // t1 = relu(hn @ pr_w1 + pr_b1) -> bf16 (vbf as [BN, 2D])
    gemm(hnbf, Dn, 0, pr_w1T, Dn, 0, pr_b1, nullptr,
         nullptr, 0, 0, nullptr, 0, 0, 1.f, 1,
         nullptr, 0, 0, 0, vbf, 2 * Dn, 1, 0, 0, BNn, 2 * Dn, Dn, 1);
    // h = t1 @ pr_w2 + pr_b2 + hn
    gemm(vbf, 2 * Dn, 0, pr_w2T, 2 * Dn, 0, pr_b2, nullptr,
         scoresF, Dn, 0, nullptr, 0, 0, 1.f, 0,
         h, Dn, 1, 0, nullptr, 0, 0, 0, 0, BNn, Dn, 2 * Dn, 1);

    // ---- node-sum + predictor head ----
    nodesum_kernel<<<(Bn * Dn + 255) / 256, 256, 0, stream>>>(h, hsum);
    head_kernel<<<Bn, 256, 0, stream>>>(hsum, sf, pd_w1, pd_b1, pd_w2, pd_b2, outp);
}
